// RNN_47321949667752
// MI455X (gfx1250) — matrix operand order repaired and checked by reading, not compiled, not run on hardware
//
#include <hip/hip_runtime.h>
#include <hip/hip_bf16.h>

// ---------------------------------------------------------------------------
// 2-layer tanh RNN, persistent batch-tiled WMMA kernel for gfx1250 (MI455X).
//   T=1024 steps, B=256, IN=128, H=256, C=2.
//   grid = 16 blocks (one 16-row batch tile each); block = 512 threads
//   (16 wave32 waves, one 16-wide N-tile of H per wave).
//   Whh0/Wih1/Whh1 live in per-wave VGPR fragments (24 frags = 192 VGPRs,
//   spill-free at 4 waves/SIMD); Wih0 lives in LDS (64 KB, copied once),
//   re-read per step via ds_load_b128 behind an opaque-offset LICM guard.
//   Each layer uses two independent WMMA accumulators to halve the serial
//   XDL dependency chain (the recurrence is latency-bound).
// ---------------------------------------------------------------------------

#define RNN_T 1024
#define RNN_B 256
#define RNN_IN 128
#define RNN_H 256
#define RNN_C 2

typedef __attribute__((ext_vector_type(16))) __bf16 v16bf;
typedef __attribute__((ext_vector_type(8)))  float  v8f;

union FragB16 {
  v16bf v;
  uint4 q[2];
};

__device__ __forceinline__ unsigned short f2bf(float f) {
  unsigned int u = __float_as_uint(f);
  u += 0x7FFFu + ((u >> 16) & 1u);   // round-to-nearest-even
  return (unsigned short)(u >> 16);
}
__device__ __forceinline__ float bf2f(unsigned short h) {
  return __uint_as_float(((unsigned int)h) << 16);
}
__device__ __forceinline__ unsigned int pack2(float a, float b) {
  return (unsigned int)f2bf(a) | ((unsigned int)f2bf(b) << 16);
}

// gfx1250 has a hardware transcendental tanh (v_tanh_f32).
__device__ __forceinline__ float fast_tanhf(float x) {
#if __has_builtin(__builtin_amdgcn_tanhf)
  return __builtin_amdgcn_tanhf(x);
#else
  return tanhf(x);
#endif
}

// A fragment (16x32 bf16, activations) from LDS, row-major with row stride RS.
// Contiguous k order (same as B): lanes<16 hold K=kt..kt+15; lanes>=16 hold K=kt+16..kt+31.
__device__ __forceinline__ v16bf load_frag_A(const unsigned short* act, int RS,
                                             int r, int hi, int kt) {
  FragB16 f;
  const unsigned short* p = act + r * RS + kt + (hi << 4);
  f.q[0] = *(const uint4*)(p);
  f.q[1] = *(const uint4*)(p + 8);
  return f.v;
}

// B fragment (32x16 bf16, weights W[n][k] row-major, row stride Kdim).
// Contiguous k order (same as A): lanes<16 hold K=kt..kt+15; lanes>=16 hold K=kt+16..kt+31.
__device__ __forceinline__ v16bf load_frag_B(const unsigned short* W, int Kdim,
                                             int r, int hi, int n0, int kt) {
  FragB16 f;
  const unsigned short* p = W + (size_t)(n0 + r) * Kdim + kt + (hi << 4);
  f.q[0] = *(const uint4*)(p);
  f.q[1] = *(const uint4*)(p + 8);
  return f.v;
}

// B fragment from LDS via plain ds_load_b128, with an opaque offset so LICM
// cannot hoist the (loop-invariant) loads out of the recurrence and blow the
// register budget. The empty volatile asm pins the offset in-loop at no cost.
__device__ __forceinline__ v16bf load_frag_B_lds(const unsigned short* Wlds,
                                                 int Kdim, int r, int hi,
                                                 int n0, int kt) {
  int off = (n0 + r) * Kdim + kt + (hi << 4);
  asm volatile("" : "+v"(off));
  FragB16 f;
  const unsigned short* p = Wlds + off;
  f.q[0] = *(const uint4*)(p);
  f.q[1] = *(const uint4*)(p + 8);
  return f.v;
}

__device__ __forceinline__ v8f wmma_bf16(v16bf a, v16bf b, v8f c) {
  return __builtin_amdgcn_wmma_f32_16x16x32_bf16(
      /*neg_a=*/false, a, /*neg_b=*/false, b,
      /*c_mod=*/(short)0, c, /*reuse_a=*/false, /*reuse_b=*/false);
}

// ---------------------------------------------------------------------------
// Prep kernel: fp32 weights -> bf16 (row-major [n][k], same as source layout),
// and fold the three per-layer bias vectors into one fp32 vector per layer.
// ws layout (bf16 elems): Wih0[32768] Whh0[65536] Wih1[65536] Whh1[65536],
// then fp32: bias0[256] bias1[256].
// ---------------------------------------------------------------------------
__global__ void rnn_prep_kernel(const float* __restrict__ Wih0,
                                const float* __restrict__ Whh0,
                                const float* __restrict__ Wih1,
                                const float* __restrict__ Whh1,
                                const float* __restrict__ bih0,
                                const float* __restrict__ bhh0,
                                const float* __restrict__ bias0,
                                const float* __restrict__ bih1,
                                const float* __restrict__ bhh1,
                                const float* __restrict__ bias1,
                                unsigned short* __restrict__ wbf,
                                float* __restrict__ biasc) {
  int i = blockIdx.x * blockDim.x + threadIdx.x;
  if (i < 32768) {
    wbf[i] = f2bf(Wih0[i]);
  } else if (i < 98304) {
    wbf[i] = f2bf(Whh0[i - 32768]);
  } else if (i < 163840) {
    wbf[i] = f2bf(Wih1[i - 98304]);
  } else if (i < 229376) {
    wbf[i] = f2bf(Whh1[i - 163840]);
  } else if (i < 229376 + 256) {
    int n = i - 229376;
    biasc[n] = bih0[n] + bhh0[n] + bias0[n];
  } else if (i < 229376 + 512) {
    int n = i - (229376 + 256);
    biasc[256 + n] = bih1[n] + bhh1[n] + bias1[n];
  }
}

// ---------------------------------------------------------------------------
// Persistent RNN kernel: each block owns a 16-row batch tile for all T steps.
// 512 threads = 16 wave32 waves; wave w owns N-tile [w*16, w*16+16) of H.
// ---------------------------------------------------------------------------
__global__ void __launch_bounds__(512, 1)
rnn_persistent_kernel(const float* __restrict__ input_seq,   // [T][B][IN] fp32
                      const float* __restrict__ hidden,      // [2][B][H]  fp32
                      const unsigned short* __restrict__ wbf,
                      const float* __restrict__ biasc,       // [2*H] fp32
                      const float* __restrict__ fc_w,        // [C][H] fp32
                      const float* __restrict__ fc_b,        // [C]    fp32
                      float* __restrict__ out) {             // logits + last_h
  const unsigned short* Wih0g = wbf;
  const unsigned short* Whh0 = wbf + 32768;
  const unsigned short* Wih1 = wbf + 98304;
  const unsigned short* Whh1 = wbf + 163840;
  const float* bias0c = biasc;
  const float* bias1c = biasc + RNN_H;

  __shared__ alignas(16) unsigned short wih0_lds[RNN_H * RNN_IN];  // 64 KB
  __shared__ alignas(16) unsigned short x_bf[16 * RNN_IN];         // 4 KB
  __shared__ alignas(16) unsigned short h0_bf[16 * RNN_H];         // 8 KB
  __shared__ alignas(16) unsigned short h1_bf[16 * RNN_H];         // 8 KB

  const int tid = threadIdx.x;
  const int bBase = blockIdx.x * 16;

  // One-time copy: Wih0 (bf16, 64 KB) global -> LDS.
  {
    const uint4* src = (const uint4*)Wih0g;
    uint4* dst = (uint4*)wih0_lds;
#pragma unroll
    for (int j = 0; j < 8; ++j) dst[tid + 512 * j] = src[tid + 512 * j];
  }

  // Init hidden state tiles from the (fp32) hidden input.
  for (int idx = tid; idx < 16 * RNN_H; idx += 512) {
    int bl = idx >> 8, h = idx & 255;
    h0_bf[bl * RNN_H + h] = f2bf(hidden[(size_t)(bBase + bl) * RNN_H + h]);
    h1_bf[bl * RNN_H + h] =
        f2bf(hidden[(size_t)RNN_B * RNN_H + (size_t)(bBase + bl) * RNN_H + h]);
  }

  const int wv = tid >> 5;        // wave id 0..15 (wave32)
  const int lane = tid & 31;
  const int r = lane & 15;
  const int hi = lane >> 4;       // 0 or 1
  const int n0 = wv * 16;         // this wave's N-tile of H

  // ---- One-time weight preload into per-wave register fragments.
  // 8 + 8 + 8 = 24 fragments x 8 VGPRs = 192 VGPRs/wave, resident for the
  // entire recurrence; fits the 256-VGPR/wave cap at 4 waves/SIMD.
  v16bf whh0[8], wih1[8], whh1[8];
#pragma unroll
  for (int i = 0; i < 8; ++i)
    whh0[i] = load_frag_B(Whh0, RNN_H, r, hi, n0, 32 * i);
#pragma unroll
  for (int i = 0; i < 8; ++i)
    wih1[i] = load_frag_B(Wih1, RNN_H, r, hi, n0, 32 * i);
#pragma unroll
  for (int i = 0; i < 8; ++i)
    whh1[i] = load_frag_B(Whh1, RNN_H, r, hi, n0, 32 * i);
  const float bias0r = bias0c[n0 + r];
  const float bias1r = bias1c[n0 + r];
  const float fcb = fc_b[(tid >> 4) & 1];  // per-thread c = bit4 of tid

  const v8f zero8 = {0.f, 0.f, 0.f, 0.f, 0.f, 0.f, 0.f, 0.f};

  __syncthreads();

  for (int t = 0; t < RNN_T; ++t) {
    // ---- Stage x_t tile into LDS as bf16 (each thread: 4 contiguous elems)
    {
      int idx4 = tid << 2;                // 2048 elems / 512 threads
      int row = idx4 >> 7;
      int col = idx4 & 127;
      const float* src =
          input_seq + ((size_t)t * RNN_B + bBase + row) * RNN_IN + col;
      float4 p = *(const float4*)(src);
      uint2 q;
      q.x = pack2(p.x, p.y);
      q.y = pack2(p.z, p.w);
      *(uint2*)(x_bf + row * RNN_IN + col) = q;
      if (t + 1 < RNN_T) {  // prefetch next step's x (global_prefetch_b8)
        __builtin_prefetch(src + (size_t)RNN_B * RNN_IN, 0, 0);
      }
    }
    __syncthreads();

    // ---- Layer 0: h0 = tanh(x @ Wih0^T + h0 @ Whh0^T + bias0c)
    // Two independent accumulators -> two parallel WMMA dependency chains.
    v8f accx = zero8, acch = zero8;
#pragma unroll
    for (int i = 0; i < 4; ++i)
      accx = wmma_bf16(load_frag_A(x_bf, RNN_IN, r, hi, 32 * i),
                       load_frag_B_lds(wih0_lds, RNN_IN, r, hi, n0, 32 * i),
                       accx);
#pragma unroll
    for (int i = 0; i < 8; ++i)
      acch = wmma_bf16(load_frag_A(h0_bf, RNN_H, r, hi, 32 * i), whh0[i], acch);
    v8f acc = accx + acch;
    __syncthreads();  // all reads of h0_bf done before overwrite
#pragma unroll
    for (int v = 0; v < 8; ++v) {
      int m = v + hi * 8;  // C/D layout: lanes<16 -> M=v, lanes>=16 -> M=v+8
      h0_bf[m * RNN_H + n0 + r] = f2bf(fast_tanhf(acc[v] + bias0r));
    }
    __syncthreads();

    // ---- Layer 1: h1 = tanh(h0 @ Wih1^T + h1 @ Whh1^T + bias1c)
    accx = zero8;
    acch = zero8;
#pragma unroll
    for (int i = 0; i < 8; ++i)
      accx = wmma_bf16(load_frag_A(h0_bf, RNN_H, r, hi, 32 * i), wih1[i], accx);
#pragma unroll
    for (int i = 0; i < 8; ++i)
      acch = wmma_bf16(load_frag_A(h1_bf, RNN_H, r, hi, 32 * i), whh1[i], acch);
    acc = accx + acch;
    __syncthreads();
#pragma unroll
    for (int v = 0; v < 8; ++v) {
      int m = v + hi * 8;
      h1_bf[m * RNN_H + n0 + r] = f2bf(fast_tanhf(acc[v] + bias1r));
    }
    __syncthreads();

    // ---- FC head: logits[t, b, c] = h1[b,:] . fc_w[c,:] + fc_b[c]
    // 32 (b,c) pairs, 16 threads each (16-elem chunks), wave32 xor-reduce
    // (masks 1,2,4,8 stay within each 16-lane half of a wave).
    {
      int p = tid >> 4;       // 0..31
      int c16 = tid & 15;     // chunk within pair
      int bl = p >> 1;
      int c = p & 1;          // == (tid >> 4) & 1
      const float* fw = fc_w + c * RNN_H + c16 * 16;
      const unsigned short* hh = h1_bf + bl * RNN_H + c16 * 16;
      float s = 0.f;
#pragma unroll
      for (int i = 0; i < 16; ++i) s += bf2f(hh[i]) * fw[i];
      s += __shfl_xor(s, 1, 32);
      s += __shfl_xor(s, 2, 32);
      s += __shfl_xor(s, 4, 32);
      s += __shfl_xor(s, 8, 32);
      if (c16 == 0) {
        out[(size_t)t * (RNN_B * RNN_C) + (size_t)(bBase + bl) * RNN_C + c] =
            s + fcb;
      }
    }
    // No barrier needed: next iter's top __syncthreads orders everything.
  }

  // ---- Final hidden state [2][B][H] (fp32) appended after logits.
  __syncthreads();
  const size_t hid_base = (size_t)RNN_T * RNN_B * RNN_C;
  for (int idx = tid; idx < 16 * RNN_H; idx += 512) {
    int bl = idx >> 8, h = idx & 255;
    out[hid_base + (size_t)(bBase + bl) * RNN_H + h] = bf2f(h0_bf[bl * RNN_H + h]);
    out[hid_base + (size_t)RNN_B * RNN_H + (size_t)(bBase + bl) * RNN_H + h] =
        bf2f(h1_bf[bl * RNN_H + h]);
  }
}

// ---------------------------------------------------------------------------
extern "C" void kernel_launch(void* const* d_in, const int* in_sizes, int n_in,
                              void* d_out, int out_size, void* d_ws,
                              size_t ws_size, hipStream_t stream) {
  const float* input_seq = (const float*)d_in[0];
  const float* hidden    = (const float*)d_in[1];
  const float* W_ih0     = (const float*)d_in[2];
  const float* b_ih0     = (const float*)d_in[3];
  const float* W_hh0     = (const float*)d_in[4];
  const float* b_hh0     = (const float*)d_in[5];
  const float* bias0     = (const float*)d_in[6];
  const float* W_ih1     = (const float*)d_in[7];
  const float* b_ih1     = (const float*)d_in[8];
  const float* W_hh1     = (const float*)d_in[9];
  const float* b_hh1     = (const float*)d_in[10];
  const float* bias1     = (const float*)d_in[11];
  const float* fc_w      = (const float*)d_in[12];
  const float* fc_b      = (const float*)d_in[13];
  float* out = (float*)d_out;

  unsigned short* wbf = (unsigned short*)d_ws;
  float* biasc = (float*)((char*)d_ws + 229376 * sizeof(unsigned short));

  const int prep_n = 229376 + 512;
  rnn_prep_kernel<<<(prep_n + 255) / 256, 256, 0, stream>>>(
      W_ih0, W_hh0, W_ih1, W_hh1, b_ih0, b_hh0, bias0, b_ih1, b_hh1, bias1,
      wbf, biasc);

  rnn_persistent_kernel<<<RNN_B / 16, 512, 0, stream>>>(
      input_seq, hidden, wbf, biasc, fc_w, fc_b, out);
}
